// HungarianMatcher_34299608826005
// MI455X (gfx1250) — matrix operand order repaired and checked by reading, not compiled, not run on hardware
//
#include <hip/hip_runtime.h>
#include <hip/hip_bf16.h>
#include <math.h>

// ---------------------------------------------------------------------------
// Problem constants (fixed by the reference)
// ---------------------------------------------------------------------------
#define BB   2
#define QQ   200
#define CC   2
#define NN   50
#define HH   256
#define WW   256
#define DSZ  64
#define PP   4096          // DSZ*DSZ
#define NP   64            // N padded to 4 WMMA tiles
#define QT   13            // ceil(Q/16)

typedef __attribute__((ext_vector_type(16))) __bf16    v16bf;
typedef __attribute__((ext_vector_type(8)))  float     v8f;
typedef __attribute__((ext_vector_type(4)))  unsigned  u32x4;

union ABfrag { v16bf v; u32x4 q[2]; };

// ---------------------------------------------------------------------------
// zero a span of 32-bit words (grid-stride)
// ---------------------------------------------------------------------------
__global__ void zero_words_kernel(unsigned* __restrict__ p, long long nwords) {
    long long i = (long long)blockIdx.x * blockDim.x + threadIdx.x;
    long long stride = (long long)gridDim.x * blockDim.x;
    for (; i < nwords; i += stride) p[i] = 0u;
}

// ---------------------------------------------------------------------------
// CDNA5 async global->LDS staging of the 8 input rows needed by one output
// row of the 4x antialiased downsample. 64 threads x 8 rows x 16B chunks.
// Tracked by ASYNCcnt; per-wave s_wait_asynccnt then workgroup barrier.
// ---------------------------------------------------------------------------
__device__ __forceinline__ void load_rows_async(const float* __restrict__ img,
                                                float (&rows)[8][WW],
                                                int orow, int tid) {
    int r0 = 4 * orow - 2;
#pragma unroll
    for (int j = 0; j < 8; ++j) {
        int ir = r0 + j;
        ir = ir < 0 ? 0 : (ir > HH - 1 ? HH - 1 : ir);
        const float* gsrc = img + (size_t)ir * WW + tid * 4;     // 16B / lane
        unsigned lds = (unsigned)(size_t)(void*)&rows[j][tid * 4];
        asm volatile("global_load_async_to_lds_b128 %0, %1, off"
                     :: "v"(lds), "v"(gsrc) : "memory");
    }
    asm volatile("s_wait_asynccnt 0x0" ::: "memory");
    __syncthreads();
}

// separable 8-tap antialiased bilinear (jax.image.resize triangle kernel):
// weights {1,3,5,7,7,5,3,1}/32, edge-clamped.
__device__ __forceinline__ float ds_pixel(const float rows[8][WW], int tid) {
    const float Wt[8] = {1.f/32.f, 3.f/32.f, 5.f/32.f, 7.f/32.f,
                         7.f/32.f, 5.f/32.f, 3.f/32.f, 1.f/32.f};
    int c0 = 4 * tid - 2;
    float acc = 0.f;
#pragma unroll
    for (int j = 0; j < 8; ++j) {
        float h = 0.f;
#pragma unroll
        for (int i = 0; i < 8; ++i) {
            int ic = c0 + i;
            ic = ic < 0 ? 0 : (ic > WW - 1 ? WW - 1 : ic);
            h += Wt[i] * rows[j][ic];
        }
        acc += Wt[j] * h;
    }
    return acc;
}

// pred path: downsample -> x ; emit bf16 panels c1,c2,c3,p and per-(q,row)
// partial sums of p^2*softplus(x) and p (reduced deterministically later)
__global__ void pred_transform_kernel(const float* __restrict__ pred_masks,
                                      __bf16* __restrict__ c1p,
                                      __bf16* __restrict__ c2p,
                                      __bf16* __restrict__ c3p,
                                      __bf16* __restrict__ pp,
                                      float* __restrict__ S0part,
                                      float* __restrict__ sumPpart) {
    __shared__ __align__(16) float rows[8][WW];
    __shared__ float red[2][64];
    int tid  = threadIdx.x;
    int orow = blockIdx.x;
    int img  = blockIdx.y;                 // b*Q + q
    int b = img / QQ, q = img % QQ;

    load_rows_async(pred_masks + (size_t)img * (HH * WW), rows, orow, tid);

    float x = ds_pixel(rows, tid);
    float p = 1.f / (1.f + __expf(-x));
    float S = fmaxf(x, 0.f) + log1pf(__expf(-fabsf(x)));
    float A0 = p * p;
    float om2p = 1.f - 2.f * p;
    float A1 = 2.f * p * om2p;
    float A2 = om2p * om2p;

    size_t o = ((size_t)(b * QQ + q)) * PP + (size_t)orow * DSZ + tid;
    c1p[o] = (__bf16)(A1 * S - A0 * x);
    c2p[o] = (__bf16)(A2 * S - A1 * x);
    c3p[o] = (__bf16)(-A2 * x);
    pp[o]  = (__bf16)p;

    red[0][tid] = A0 * S;
    red[1][tid] = p;
    __syncthreads();
#pragma unroll
    for (int s = 32; s > 0; s >>= 1) {
        if (tid < s) { red[0][tid] += red[0][tid + s]; red[1][tid] += red[1][tid + s]; }
        __syncthreads();
    }
    if (tid == 0) {
        S0part  [(size_t)img * DSZ + orow] = red[0][0];
        sumPpart[(size_t)img * DSZ + orow] = red[1][0];
    }
}

// gt path: downsample -> g ; emit bf16 panels g, g^2, g^3 (N padded to NP,
// pad rows pre-zeroed) and per-(n,row) partial sum of g
__global__ void gt_transform_kernel(const float* __restrict__ gt_masks,
                                    __bf16* __restrict__ gp,
                                    __bf16* __restrict__ g2p,
                                    __bf16* __restrict__ g3p,
                                    float* __restrict__ sumGpart) {
    __shared__ __align__(16) float rows[8][WW];
    __shared__ float red[64];
    int tid  = threadIdx.x;
    int orow = blockIdx.x;
    int img  = blockIdx.y;                 // b*N + n
    int b = img / NN, n = img % NN;

    load_rows_async(gt_masks + (size_t)img * (HH * WW), rows, orow, tid);

    float g = ds_pixel(rows, tid);
    size_t o = ((size_t)(b * NP + n)) * PP + (size_t)orow * DSZ + tid;
    gp[o]  = (__bf16)g;
    g2p[o] = (__bf16)(g * g);
    g3p[o] = (__bf16)(g * g * g);

    red[tid] = g;
    __syncthreads();
#pragma unroll
    for (int s = 32; s > 0; s >>= 1) {
        if (tid < s) red[tid] += red[tid + s];
        __syncthreads();
    }
    if (tid == 0) sumGpart[(size_t)img * DSZ + orow] = red[0];
}

// deterministic 64-wide reduction of per-row partials -> per-image scalars
__global__ void reduce64_kernel(const float* __restrict__ in,
                                float* __restrict__ out) {
    __shared__ float red[64];
    int tid = threadIdx.x;
    int i = blockIdx.x;
    red[tid] = in[(size_t)i * 64 + tid];
    __syncthreads();
#pragma unroll
    for (int s = 32; s > 0; s >>= 1) {
        if (tid < s) red[tid] += red[tid + s];
        __syncthreads();
    }
    if (tid == 0) out[i] = red[0];
}

// ---------------------------------------------------------------------------
// Fused WMMA cost kernel: 4 bf16 GEMMs (c1*g, c2*g^2, c3*g^3, p*g) over
// K = 4096, tiled 16x16x32, one block = 16 q-rows, one wave = one 16-n tile.
// Software-pipelined: two disjoint fragment register sets so the 14 loads of
// chunk k+32 overlap the 4 WMMAs of chunk k (partial s_wait_loadcnt instead
// of wait-0 serialization). Panels carry one trailing pad row so the final
// dead prefetch stays in-bounds.
// ---------------------------------------------------------------------------
__device__ __forceinline__ void load_frags(ABfrag* f,
                                           const __bf16* __restrict__ c1A,
                                           const __bf16* __restrict__ c2A,
                                           const __bf16* __restrict__ c3A,
                                           const __bf16* __restrict__ pA,
                                           const __bf16* __restrict__ gB,
                                           const __bf16* __restrict__ g2B,
                                           const __bf16* __restrict__ g3B,
                                           size_t abase, size_t bbase, int k0) {
    size_t ao = abase + k0, bo = bbase + k0;
    f[0].q[0] = *(const u32x4*)(c1A + ao);  f[0].q[1] = *(const u32x4*)(c1A + ao + 16);
    f[1].q[0] = *(const u32x4*)(c2A + ao);  f[1].q[1] = *(const u32x4*)(c2A + ao + 16);
    f[2].q[0] = *(const u32x4*)(c3A + ao);  f[2].q[1] = *(const u32x4*)(c3A + ao + 16);
    f[3].q[0] = *(const u32x4*)(pA  + ao);  f[3].q[1] = *(const u32x4*)(pA  + ao + 16);
    f[4].q[0] = *(const u32x4*)(gB  + bo);  f[4].q[1] = *(const u32x4*)(gB  + bo + 16);
    f[5].q[0] = *(const u32x4*)(g2B + bo);  f[5].q[1] = *(const u32x4*)(g2B + bo + 16);
    f[6].q[0] = *(const u32x4*)(g3B + bo);  f[6].q[1] = *(const u32x4*)(g3B + bo + 16);
}

__device__ __forceinline__ void do_wmmas(const ABfrag* f, v8f& a1, v8f& a2,
                                         v8f& a3, v8f& aI) {
    a1 = __builtin_amdgcn_wmma_f32_16x16x32_bf16(false, f[0].v, false, f[4].v,
                                                 (short)0, a1, false, false);
    a2 = __builtin_amdgcn_wmma_f32_16x16x32_bf16(false, f[1].v, false, f[5].v,
                                                 (short)0, a2, false, false);
    a3 = __builtin_amdgcn_wmma_f32_16x16x32_bf16(false, f[2].v, false, f[6].v,
                                                 (short)0, a3, false, false);
    aI = __builtin_amdgcn_wmma_f32_16x16x32_bf16(false, f[3].v, false, f[4].v,
                                                 (short)0, aI, false, false);
}

__global__ void wmma_cost_kernel(const __bf16* __restrict__ c1A,
                                 const __bf16* __restrict__ c2A,
                                 const __bf16* __restrict__ c3A,
                                 const __bf16* __restrict__ pA,
                                 const __bf16* __restrict__ gB,
                                 const __bf16* __restrict__ g2B,
                                 const __bf16* __restrict__ g3B,
                                 const float* __restrict__ S0,
                                 const float* __restrict__ sumP,
                                 const float* __restrict__ sumG,
                                 const float* __restrict__ logits,
                                 const int*   __restrict__ labels,
                                 float* __restrict__ out) {
    int b    = blockIdx.x / QT;
    int qt   = blockIdx.x % QT;
    int wave = threadIdx.x >> 5;
    int lane = threadIdx.x & 31;
    int n0   = wave * 16;

    // A fragment (16x32 bf16): lane-half kb in {0,8}; elems 0-7 <- K kb..kb+7,
    // elems 8-15 <- K kb+16..kb+23  (two contiguous 16B loads per lane)
    int mrow = qt * 16 + (lane & 15);
    if (mrow > QQ - 1) mrow = QQ - 1;          // clamped; edge rows masked at store
    size_t abase = ((size_t)(b * QQ + mrow)) * PP + (size_t)((lane >> 4) * 8);

    // B fragment (32x16 bf16): lane-half kb in {0,8}; elems 0-7 <- K kb..kb+7,
    // elems 8-15 <- K kb+16..kb+23 (same order as A)
    int nrow = n0 + (lane & 15);               // pad rows pre-zeroed
    size_t bbase = ((size_t)(b * NP + nrow)) * PP + (size_t)((lane >> 4) * 8);

    v8f acc1 = {}, acc2 = {}, acc3 = {}, accI = {};
    ABfrag s1[7], s2[7];

    load_frags(s1, c1A, c2A, c3A, pA, gB, g2B, g3B, abase, bbase, 0);
    for (int k0 = 0; k0 < PP; k0 += 64) {
        load_frags(s2, c1A, c2A, c3A, pA, gB, g2B, g3B, abase, bbase, k0 + 32);
        do_wmmas(s1, acc1, acc2, acc3, accI);
        load_frags(s1, c1A, c2A, c3A, pA, gB, g2B, g3B, abase, bbase, k0 + 64);
        do_wmmas(s2, acc1, acc2, acc3, accI);   // last s1 prefetch is dead (pad row)
    }

    // Epilogue. C/D layout: VGPR r -> (M = r + 8*(lane>=16), N = lane&15)
    int nn = n0 + (lane & 15);
#pragma unroll
    for (int r = 0; r < 8; ++r) {
        int m = qt * 16 + r + (lane >> 4) * 8;
        if (m < QQ && nn < NN) {
            float s0v = S0[b * QQ + m];
            float spv = sumP[b * QQ + m];
            float sgv = sumG[b * NN + nn];
            float cmask = (0.25f / (float)PP) * (s0v + acc1[r] + acc2[r] + acc3[r]);
            float dice  = 1.f - (2.f * accI[r] + 1.f) / (spv + sgv + 1.f);
            float l0 = logits[((size_t)(b * QQ + m)) * CC + 0];
            float l1 = logits[((size_t)(b * QQ + m)) * CC + 1];
            float mx = fmaxf(l0, l1);
            float e0 = __expf(l0 - mx), e1 = __expf(l1 - mx);
            int lab = labels[b * NN + nn];
            float prob = (lab == 0 ? e0 : e1) / (e0 + e1);
            out[((size_t)(b * QQ + m)) * NN + nn] = -prob + cmask + dice;
        }
    }
}

// ---------------------------------------------------------------------------
// launcher
// ---------------------------------------------------------------------------
extern "C" void kernel_launch(void* const* d_in, const int* in_sizes, int n_in,
                              void* d_out, int out_size, void* d_ws, size_t ws_size,
                              hipStream_t stream) {
    const float* pred_logits = (const float*)d_in[0];   // (B,Q,C)
    const float* pred_masks  = (const float*)d_in[1];   // (B,Q,H,W)
    const int*   gt_labels   = (const int*)  d_in[2];   // (B,N)
    const float* gt_masks    = (const float*)d_in[3];   // (B,N,H,W)
    float*       out         = (float*)d_out;           // (B,Q,N)

    char* ws = (char*)d_ws;
    size_t off = 0;
    auto carve = [&](size_t bytes) -> size_t {
        size_t o = off;
        off = (off + bytes + 255) & ~(size_t)255;
        return o;
    };
    // +1 trailing pad row per panel: the pipelined GEMM's final dead prefetch
    // reads up to 32 elements past the last real row.
    const size_t aPanel = (size_t)(BB * QQ + 1) * PP * sizeof(__bf16);
    const size_t bPanel = (size_t)(BB * NP + 1) * PP * sizeof(__bf16);

    size_t c1Off  = carve(aPanel);
    size_t c2Off  = carve(aPanel);
    size_t c3Off  = carve(aPanel);
    size_t pOff   = carve(aPanel);
    size_t gOff   = carve(bPanel);
    size_t g2Off  = carve(bPanel);
    size_t g3Off  = carve(bPanel);
    size_t gEnd   = off;
    size_t s0pOff = carve((size_t)BB * QQ * DSZ * sizeof(float));
    size_t sppOff = carve((size_t)BB * QQ * DSZ * sizeof(float));
    size_t sgpOff = carve((size_t)BB * NN * DSZ * sizeof(float));
    size_t s0Off  = carve((size_t)BB * QQ * sizeof(float));
    size_t spOff  = carve((size_t)BB * QQ * sizeof(float));
    size_t sgOff  = carve((size_t)BB * NN * sizeof(float));
    (void)ws_size;

    __bf16* c1p = (__bf16*)(ws + c1Off);
    __bf16* c2p = (__bf16*)(ws + c2Off);
    __bf16* c3p = (__bf16*)(ws + c3Off);
    __bf16* pp  = (__bf16*)(ws + pOff);
    __bf16* gp  = (__bf16*)(ws + gOff);
    __bf16* g2p = (__bf16*)(ws + g2Off);
    __bf16* g3p = (__bf16*)(ws + g3Off);
    float* S0part   = (float*)(ws + s0pOff);
    float* sumPpart = (float*)(ws + sppOff);
    float* sumGpart = (float*)(ws + sgpOff);
    float* S0   = (float*)(ws + s0Off);
    float* sumP = (float*)(ws + spOff);
    float* sumG = (float*)(ws + sgOff);

    // zero the g panels so the N padding rows read as 0 in the GEMM
    long long zwords = (long long)(gEnd - gOff) / 4;
    zero_words_kernel<<<512, 256, 0, stream>>>((unsigned*)(ws + gOff), zwords);

    pred_transform_kernel<<<dim3(DSZ, BB * QQ), 64, 0, stream>>>(
        pred_masks, c1p, c2p, c3p, pp, S0part, sumPpart);
    gt_transform_kernel<<<dim3(DSZ, BB * NN), 64, 0, stream>>>(
        gt_masks, gp, g2p, g3p, sumGpart);

    reduce64_kernel<<<BB * QQ, 64, 0, stream>>>(S0part, S0);
    reduce64_kernel<<<BB * QQ, 64, 0, stream>>>(sumPpart, sumP);
    reduce64_kernel<<<BB * NN, 64, 0, stream>>>(sumGpart, sumG);

    wmma_cost_kernel<<<BB * QT, 128, 0, stream>>>(
        c1p, c2p, c3p, pp, gp, g2p, g3p, S0, sumP, sumG,
        pred_logits, gt_labels, out);
}
